// PATMultiHeadAttention_23244363006066
// MI455X (gfx1250) — matrix operand order repaired and checked by reading, not compiled, not run on hardware
//
#include <hip/hip_runtime.h>

// PAT multi-head attention for MI455X (gfx1250, wave32, WMMA).
// B=16, S=1024, D=256 (per-head dim == D), H=12, P=H*D=3072.
#define BB 16
#define SS 1024
#define DD 256
#define HH 12
#define PP 3072

typedef __bf16 bf16_t;
typedef __attribute__((ext_vector_type(16))) __bf16 v16bf;
typedef __attribute__((ext_vector_type(8)))  __bf16 v8bf;
typedef __attribute__((ext_vector_type(8)))  float   v8f;
typedef int vi4 __attribute__((vector_size(16)));   // matches builtin param pointee

#if __has_builtin(__builtin_amdgcn_global_load_async_to_lds_b128)
#define HAVE_ASYNC_LDS 1
#else
#define HAVE_ASYNC_LDS 0
#endif

#if HAVE_ASYNC_LDS
# if __has_builtin(__builtin_amdgcn_s_wait_asynccnt)
#  define WAIT_ASYNC(n) __builtin_amdgcn_s_wait_asynccnt(n)
# else
#  define WAIT_ASYNC(n) asm volatile("s_wait_asynccnt %0" :: "i"(n) : "memory")
# endif
#else
# define WAIT_ASYNC(n) ((void)0)
#endif

static __device__ __forceinline__ bf16_t f2bf(float f) {
    unsigned u = __builtin_bit_cast(unsigned, f);
    unsigned r = (u + 0x7FFFu + ((u >> 16) & 1u)) >> 16;   // RNE
    unsigned short s = (unsigned short)r;
    return __builtin_bit_cast(bf16_t, s);
}

// Async (ASYNCcnt-tracked) 16B global->LDS copy per lane; sync fallback.
static __device__ __forceinline__ void async_cp16(const bf16_t* g, bf16_t* l) {
#if HAVE_ASYNC_LDS
    __builtin_amdgcn_global_load_async_to_lds_b128(
        (__attribute__((address_space(1))) vi4*)g,
        (__attribute__((address_space(3))) vi4*)l, 0, 0);
#else
    *reinterpret_cast<uint4*>(l) = *reinterpret_cast<const uint4*>(g);
#endif
}

// A-fragment (16x32 bf16, M x K). ISA layout: lanes 0-15 hold K {0-7,16-23},
// lanes 16-31 hold K {8-15,24-31}  -> two 16B loads per lane.
static __device__ __forceinline__ v16bf load_a_frag(const bf16_t* base, int stride) {
    const int l = threadIdx.x & 31;
    const int r = l & 15;      // M row
    const int h = l >> 4;      // lane half
    const bf16_t* p = base + (long)r * stride + h * 8;
    v8bf lo = *reinterpret_cast<const v8bf*>(p);       // K = h*8 + 0..7
    v8bf hi = *reinterpret_cast<const v8bf*>(p + 16);  // K = 16 + h*8 + 0..7
    return __builtin_shufflevector(lo, hi, 0,1,2,3,4,5,6,7,8,9,10,11,12,13,14,15);
}

// B-fragment (32x16 bf16, K x N) loaded from B^T stored row-major (N rows, K contig).
// ISA layout: lane = N column; half0 K {0-7,16-23}, half1 K {8-15,24-31} -> two 16B loads per lane.
static __device__ __forceinline__ v16bf load_b_frag(const bf16_t* base, int stride) {
    const int l = threadIdx.x & 31;
    const bf16_t* p = base + (long)(l & 15) * stride + (l >> 4) * 8;
    v8bf lo = *reinterpret_cast<const v8bf*>(p);
    v8bf hi = *reinterpret_cast<const v8bf*>(p + 16);
    return __builtin_shufflevector(lo, hi, 0,1,2,3,4,5,6,7,8,9,10,11,12,13,14,15);
}

static __device__ __forceinline__ v8f wmma_bf16(v16bf a, v16bf b, v8f c) {
    return __builtin_amdgcn_wmma_f32_16x16x32_bf16(false, a, false, b, (short)0, c, false, false);
}

// ---------------------------------------------------------------- converts
__global__ void k_cvt_bf16(const float* __restrict__ in, bf16_t* __restrict__ out, int n) {
    int i = blockIdx.x * blockDim.x + threadIdx.x;
    if (i < n) out[i] = f2bf(in[i]);
}

// out[c*R + r] = in[r*C + c]  (bf16), i.e. store W^T with K contiguous per row.
__global__ void k_transpose_bf16(const float* __restrict__ in, bf16_t* __restrict__ out,
                                 int R, int C) {
    int i = blockIdx.x * blockDim.x + threadIdx.x;
    if (i >= R * C) return;
    int r = i / C, c = i - r * C;
    out[(long)c * R + r] = f2bf(in[(long)r * C + c]);
}

// ---------------------------------------------------------------- QKV GEMM
// C[16384,3072] = X[16384,256] @ W[256,3072] + bias.  One 64x64 tile per wave.
// Q/K stored [B,H,S,D] (d contig); V stored transposed [B,H,D,S] (s contig).
__global__ __launch_bounds__(128) void k_qkv_gemm(const bf16_t* __restrict__ X,
                                                  const bf16_t* __restrict__ Wt,   // [P,D]
                                                  const float*  __restrict__ bias, // [P]
                                                  bf16_t* __restrict__ out,
                                                  int vtrans) {
    const int lane = threadIdx.x & 31;
    const int wg   = blockIdx.x * 4 + (threadIdx.x >> 5);   // 12288 waves
    const int NT   = PP / 64;                                // 48
    const int nt   = wg % NT;
    const int mt   = wg / NT;                                // 0..255
    const int m0 = mt * 64, n0 = nt * 64;

    const v8f zero = {0.f,0.f,0.f,0.f,0.f,0.f,0.f,0.f};
    v8f acc[4][4];
#pragma unroll
    for (int i = 0; i < 4; ++i)
#pragma unroll
        for (int j = 0; j < 4; ++j) acc[i][j] = zero;

    for (int dc = 0; dc < DD; dc += 32) {
        v16bf a[4], b[4];
#pragma unroll
        for (int i = 0; i < 4; ++i) a[i] = load_a_frag(X + (long)(m0 + i*16) * DD + dc, DD);
#pragma unroll
        for (int j = 0; j < 4; ++j) b[j] = load_b_frag(Wt + (long)(n0 + j*16) * DD + dc, DD);
#pragma unroll
        for (int i = 0; i < 4; ++i)
#pragma unroll
            for (int j = 0; j < 4; ++j) acc[i][j] = wmma_bf16(a[i], b[j], acc[i][j]);
    }

    const int h = lane >> 4, nl = lane & 15;
    float bj[4];
#pragma unroll
    for (int j = 0; j < 4; ++j) bj[j] = bias[n0 + j*16 + nl];

#pragma unroll
    for (int i = 0; i < 4; ++i)
#pragma unroll
        for (int j = 0; j < 4; ++j)
#pragma unroll
            for (int r = 0; r < 8; ++r) {
                float v = acc[i][j][r] + bj[j];
                int m = m0 + i*16 + r + 8*h;     // b*S + s
                int n = n0 + j*16 + nl;          // head*D + d
                int head = n >> 8, d = n & 255;
                int b_ = m >> 10, s = m & 1023;
                long idx = vtrans
                    ? (((long)(b_*HH + head) * DD + d) * SS + s)
                    : (((long)(b_*HH + head) * SS + s) * DD + d);
                out[idx] = f2bf(v);
            }
}

// ---------------------------------------------------------------- flash attention
// Block = 4 waves sharing one (b,head); each wave owns 16 query rows.
// Per 32-key block: K tile (32x256, 16KB) and V^T tile (256x32, 16KB) are staged
// into LDS via async global->LDS DMA, double-buffered so the next tile's copy
// overlaps the current tile's 32 WMMAs.  Scores computed TRANSPOSED
// (S^T = K @ Q^T) so the C-layout of the exp'd tile is exactly the A-fragment
// layout for the P@V WMMA (no cross-lane transpose).
__global__ __launch_bounds__(128) void k_flash_attn(const bf16_t* __restrict__ Q,
                                                    const bf16_t* __restrict__ K,
                                                    const bf16_t* __restrict__ Vt,
                                                    bf16_t* __restrict__ AO) {
    __shared__ __attribute__((aligned(32))) bf16_t skA[2][32 * DD];  // K rows, stride DD
    __shared__ __attribute__((aligned(32))) bf16_t svA[2][DD * 32];  // V^T rows d, stride 32

    const int tid  = threadIdx.x;
    const int lane = tid & 31;
    const int wave = tid >> 5;
    const int blk  = blockIdx.x;           // 3072 blocks = 16b x 12h x 16 q-blocks
    const int qblk = blk & 15;
    const int head = (blk >> 4) % HH;
    const int b    = blk / (16 * HH);
    const int bh   = b * HH + head;
    const int q0   = qblk * 64 + wave * 16;
    const int h    = lane >> 4;

    const bf16_t* Qb = Q  + ((long)bh * SS + q0) * DD;
    const bf16_t* Kb = K  + (long)bh * SS * DD;
    const bf16_t* Vb = Vt + (long)bh * DD * SS;

    // Q tile (16 x 256) preloaded as B operands, 8 d-chunks.
    v16bf qf[8];
#pragma unroll
    for (int dc = 0; dc < 8; ++dc) qf[dc] = load_b_frag(Qb + dc * 32, DD);

    const v8f zero = {0.f,0.f,0.f,0.f,0.f,0.f,0.f,0.f};
    v8f O[16];
#pragma unroll
    for (int nb = 0; nb < 16; ++nb) O[nb] = zero;

    float mrun = -3.0e38f, lrun = 0.f;
    const float scale = 0.0625f;   // 1/sqrt(256)

    // Stage one 32-key block: 16 async 16B copies per thread (8 K + 8 V).
    auto stage = [&](int buf, int kb) {
        const bf16_t* gk = Kb + (long)kb * DD;    // 16KB contiguous
        bf16_t* lk = skA[buf];
        bf16_t* lv = svA[buf];
#pragma unroll
        for (int i = 0; i < 8; ++i) {
            int c = tid + i * 128;                // 1024 chunks of 8 elems
            async_cp16(gk + c * 8, lk + c * 8);
        }
#pragma unroll
        for (int i = 0; i < 8; ++i) {
            int c = tid + i * 128;
            int d = c >> 2, part = (c & 3) * 8;   // 256 rows x 4 chunks
            async_cp16(Vb + (long)d * SS + kb + part, lv + d * 32 + part);
        }
    };

    stage(0, 0);

    for (int it = 0; it < SS / 32; ++it) {
        const int cur = it & 1;
        if (it + 1 < SS / 32) {
            stage(cur ^ 1, (it + 1) * 32);
            WAIT_ASYNC(16);      // async ops retire in order: current buffer landed
        } else {
            WAIT_ASYNC(0);
        }
        __syncthreads();

        const bf16_t* lk = skA[cur];
        const bf16_t* lv = svA[cur];

        // S^T block [32 keys x 16 q] as two 16x16 WMMA tiles, fed from LDS.
        v8f st0 = zero, st1 = zero;
#pragma unroll
        for (int dc = 0; dc < 8; ++dc) {
            v16bf ka0 = load_a_frag(lk + dc * 32, DD);
            v16bf ka1 = load_a_frag(lk + 16 * DD + dc * 32, DD);
            st0 = wmma_bf16(ka0, qf[dc], st0);
            st1 = wmma_bf16(ka1, qf[dc], st1);
        }

        // Online softmax: lanes l and l^16 each hold 16 of the 32 keys of q = lane&15.
        float tmax = -3.0e38f;
#pragma unroll
        for (int r = 0; r < 8; ++r) {
            st0[r] *= scale; st1[r] *= scale;
            tmax = fmaxf(tmax, fmaxf(st0[r], st1[r]));
        }
        tmax = fmaxf(tmax, __shfl_xor(tmax, 16, 32));
        float mnew  = fmaxf(mrun, tmax);
        float alpha = __expf(mrun - mnew);
        float rs = 0.f;
        v16bf pa;   // P tile packed directly in A-fragment order
#pragma unroll
        for (int r = 0; r < 8; ++r) {
            float p0 = __expf(st0[r] - mnew);
            float p1 = __expf(st1[r] - mnew);
            rs += p0 + p1;
            pa[r]     = f2bf(p0);   // k = r + 8h        (A VGPRs 0-3)
            pa[r + 8] = f2bf(p1);   // k = 16 + r + 8h   (A VGPRs 4-7)
        }
        rs += __shfl_xor(rs, 16, 32);
        lrun = lrun * alpha + rs;
        mrun = mnew;

        // Broadcast alpha to the O C-layout rows (row q = r + 8h).
        float arow[8];
#pragma unroll
        for (int r = 0; r < 8; ++r) arow[r] = __shfl(alpha, r + (h << 3), 32);

#pragma unroll
        for (int nb = 0; nb < 16; ++nb) {
            v16bf vb = load_b_frag(lv + (nb * 16) * 32, 32);   // V^T rows = d
            v8f o = O[nb];
#pragma unroll
            for (int r = 0; r < 8; ++r) o[r] *= arow[r];
            O[nb] = wmma_bf16(pa, vb, o);
        }

        __syncthreads();   // protect the buffer the next iteration overwrites
    }

    float il = 1.0f / lrun;
    float linv[8];
#pragma unroll
    for (int r = 0; r < 8; ++r) linv[r] = __shfl(il, r + (h << 3), 32);

    const int nl = lane & 15;
    bf16_t* aob = AO + ((long)b * SS + q0) * PP + head * DD;
#pragma unroll
    for (int nb = 0; nb < 16; ++nb)
#pragma unroll
        for (int r = 0; r < 8; ++r) {
            float o = O[nb][r] * linv[r];
            aob[(long)(r + 8*h) * PP + nb * 16 + nl] = f2bf(o);
        }
}

// ---------------------------------------------------------------- output GEMM
// out[16384,256] = AO[16384,3072] @ Wo[3072,256] + bo   (fp32 out)
__global__ __launch_bounds__(128) void k_out_gemm(const bf16_t* __restrict__ AO,
                                                  const bf16_t* __restrict__ Wot,  // [D,P]
                                                  const float*  __restrict__ bo,
                                                  float* __restrict__ out) {
    const int lane = threadIdx.x & 31;
    const int wg   = blockIdx.x * 4 + (threadIdx.x >> 5);   // 1024 waves
    const int nt = wg & 3;           // 4 n-tiles
    const int mt = wg >> 2;          // 256 m-tiles
    const int m0 = mt * 64, n0 = nt * 64;

    const v8f zero = {0.f,0.f,0.f,0.f,0.f,0.f,0.f,0.f};
    v8f acc[4][4];
#pragma unroll
    for (int i = 0; i < 4; ++i)
#pragma unroll
        for (int j = 0; j < 4; ++j) acc[i][j] = zero;

    for (int dc = 0; dc < PP; dc += 32) {
        v16bf a[4], b[4];
#pragma unroll
        for (int i = 0; i < 4; ++i) a[i] = load_a_frag(AO + (long)(m0 + i*16) * PP + dc, PP);
#pragma unroll
        for (int j = 0; j < 4; ++j) b[j] = load_b_frag(Wot + (long)(n0 + j*16) * PP + dc, PP);
#pragma unroll
        for (int i = 0; i < 4; ++i)
#pragma unroll
            for (int j = 0; j < 4; ++j) acc[i][j] = wmma_bf16(a[i], b[j], acc[i][j]);
    }

    const int h = lane >> 4, nl = lane & 15;
    float bj[4];
#pragma unroll
    for (int j = 0; j < 4; ++j) bj[j] = bo[n0 + j*16 + nl];
#pragma unroll
    for (int i = 0; i < 4; ++i)
#pragma unroll
        for (int j = 0; j < 4; ++j)
#pragma unroll
            for (int r = 0; r < 8; ++r) {
                int m = m0 + i*16 + r + 8*h;
                int n = n0 + j*16 + nl;
                out[(long)m * DD + n] = acc[i][j][r] + bj[j];
            }
}

// ---------------------------------------------------------------- host
extern "C" void kernel_launch(void* const* d_in, const int* in_sizes, int n_in,
                              void* d_out, int out_size, void* d_ws, size_t ws_size,
                              hipStream_t stream) {
    const float* x  = (const float*)d_in[0];
    const float* Wq = (const float*)d_in[1];
    const float* bq = (const float*)d_in[2];
    const float* Wk = (const float*)d_in[3];
    const float* bk = (const float*)d_in[4];
    const float* Wv = (const float*)d_in[5];
    const float* bv = (const float*)d_in[6];
    const float* Wo = (const float*)d_in[7];
    const float* bo = (const float*)d_in[8];
    float* out = (float*)d_out;

    char* ws = (char*)d_ws;
    size_t off = 0;
    auto carve = [&](size_t bytes) -> char* {
        char* p = ws + off;
        off += (bytes + 255) & ~(size_t)255;
        return p;
    };
    bf16_t* xb  = (bf16_t*)carve((size_t)BB * SS * DD * 2);     //   8 MB
    bf16_t* wqt = (bf16_t*)carve((size_t)PP * DD * 2);          // 1.5 MB
    bf16_t* wkt = (bf16_t*)carve((size_t)PP * DD * 2);
    bf16_t* wvt = (bf16_t*)carve((size_t)PP * DD * 2);
    bf16_t* wot = (bf16_t*)carve((size_t)DD * PP * 2);
    bf16_t* Qb  = (bf16_t*)carve((size_t)BB * HH * SS * DD * 2); // 96 MB
    bf16_t* Kb  = (bf16_t*)carve((size_t)BB * HH * SS * DD * 2);
    bf16_t* Vtb = (bf16_t*)carve((size_t)BB * HH * DD * SS * 2);
    bf16_t* AOb = (bf16_t*)carve((size_t)BB * SS * PP * 2);
    (void)ws_size; (void)in_sizes; (void)n_in; (void)out_size;   // ~417 MB total scratch

    const int nx = BB * SS * DD;       // 4,194,304
    k_cvt_bf16<<<(nx + 255) / 256, 256, 0, stream>>>(x, xb, nx);

    const int nw = DD * PP;            // 786,432
    k_transpose_bf16<<<(nw + 255) / 256, 256, 0, stream>>>(Wq, wqt, DD, PP);
    k_transpose_bf16<<<(nw + 255) / 256, 256, 0, stream>>>(Wk, wkt, DD, PP);
    k_transpose_bf16<<<(nw + 255) / 256, 256, 0, stream>>>(Wv, wvt, DD, PP);
    k_transpose_bf16<<<(nw + 255) / 256, 256, 0, stream>>>(Wo, wot, PP, DD);

    // 12288 waves each: 256 m-tiles x 48 n-tiles of 64x64
    k_qkv_gemm<<<3072, 128, 0, stream>>>(xb, wqt, bq, Qb, 0);
    k_qkv_gemm<<<3072, 128, 0, stream>>>(xb, wkt, bk, Kb, 0);
    k_qkv_gemm<<<3072, 128, 0, stream>>>(xb, wvt, bv, Vtb, 1);  // V stored transposed

    // 3072 blocks x 128: 16b x 12h x 16 q-blocks, async double-buffered K/V in LDS
    k_flash_attn<<<3072, 128, 0, stream>>>(Qb, Kb, Vtb, AOb);

    // 1024 waves: 256 m-tiles x 4 n-tiles
    k_out_gemm<<<256, 128, 0, stream>>>(AOb, wot, bo, out);
}
